// MultiHeadSelfAttention_11252814315611
// MI455X (gfx1250) — matrix operand order repaired and checked by reading, not compiled, not run on hardware
//
#include <hip/hip_runtime.h>
#include <hip/hip_bf16.h>

typedef __attribute__((ext_vector_type(16))) _Float16 v16h;
typedef __attribute__((ext_vector_type(8)))  _Float16 v8h;
typedef __attribute__((ext_vector_type(8)))  float    v8f;
typedef int v4i __attribute__((vector_size(16)));   // matches builtin's expected pointee

#define EMBED  1024
#define NHEAD  16
#define HDIM   64
#define BATCH  2
#define SEQ    2048
#define ROWS   (BATCH * SEQ)   // 4096

#define LDSPAD 40              // padded row stride (halves): bank step 20 mod 64 -> conflict free
#define KPAD   72              // K-tile row stride (halves): bank step 36 mod 64 -> conflict free

// ---- CDNA5 async global->LDS copy (ASYNCcnt-tracked), probe-guarded --------
#if defined(__has_builtin)
#  if __has_builtin(__builtin_amdgcn_global_load_async_to_lds_b128) && \
      __has_builtin(__builtin_amdgcn_s_wait_asynccnt)
#    define USE_ASYNC 1
#  endif
#endif
#ifndef USE_ASYNC
#  define USE_ASYNC 0
#endif

#if USE_ASYNC
// signature (from frontend diagnostic): (v4i as1*, v4i as3*, imm offset, imm cpol);
// the same immediate offset is applied to both LDS and global addresses (ISA §8.4.4)
#define ASYNC_B128(g, l, OFF)                                               \
    __builtin_amdgcn_global_load_async_to_lds_b128(                         \
        (__attribute__((address_space(1))) v4i*)(void*)(g),                 \
        (__attribute__((address_space(3))) v4i*)(void*)(l), (OFF), 0)
#define ASYNC_WAIT() __builtin_amdgcn_s_wait_asynccnt(0)
#else
#define ASYNC_WAIT()
#endif

// ---------------------------------------------------------------------------
// CDNA5 WMMA 16x16x32 f16 fragment layouts (ISA 7.12.2):
//  A (16x32): lane L (half=L>>4, m=L&15): e=0..7 -> k=half*8+e ; e=8..15 -> k=16+half*8+(e-8)
//  B (32x16): lane L (half=L>>4, n=L&15): e=0..7 -> k=half*8+e ; e=8..15 -> k=16+half*8+(e-8)  (same as A)
//  C/D f32 : lane L holds col n=L&15, element r -> row 8*half + r
// ---------------------------------------------------------------------------

__device__ __forceinline__ v16h cat8(v8h lo, v8h hi) {
    return __builtin_shufflevector(lo, hi, 0,1,2,3,4,5,6,7,8,9,10,11,12,13,14,15);
}
__device__ __forceinline__ v16h lds_read_a(const _Float16* p) {   // runs at +0 and +16
    return cat8(*(const v8h*)p, *(const v8h*)(p + 16));
}
__device__ __forceinline__ v16h lds_read_b(const _Float16* p) {   // runs at +0 and +16
    return cat8(*(const v8h*)p, *(const v8h*)(p + 16));
}
__device__ __forceinline__ v8f zero8() {
    v8f c;
#pragma unroll
    for (int i = 0; i < 8; ++i) c[i] = 0.0f;
    return c;
}
__device__ __forceinline__ v8h cvt8(float4 a, float4 b) {
    v8h h;
    h[0]=(_Float16)a.x; h[1]=(_Float16)a.y; h[2]=(_Float16)a.z; h[3]=(_Float16)a.w;
    h[4]=(_Float16)b.x; h[5]=(_Float16)b.y; h[6]=(_Float16)b.z; h[7]=(_Float16)b.w;
    return h;
}

// ---------------------------------------------------------------------------
// Stage 1: Q/K/V projection.  C = X(f32) @ W(f32) + b  -> f16 [B,H,S,Dh].
// 64x64 block tile, 2x2 waves, 32x32 per wave, double-buffered LDS staging.
// ---------------------------------------------------------------------------
__global__ __launch_bounds__(128) void qkv_gemm(
    const float* __restrict__ X, const float* __restrict__ W,
    const float* __restrict__ bias, _Float16* __restrict__ Out) {
    __shared__ __align__(16) _Float16 As[2][64 * LDSPAD];   // [m][k]
    __shared__ __align__(16) _Float16 Bs[2][64 * LDSPAD];   // [n][k]

    const int lane = threadIdx.x, wy = threadIdx.y;
    const int half = lane >> 4, nl = lane & 15;
    const int wm = wy >> 1, wn = wy & 1;
    const int row0 = blockIdx.y * 64, n0 = blockIdx.x * 64;
    const int tid = wy * 32 + lane;

    const int arow = tid >> 1, akoff = (tid & 1) * 16;   // A: 16 floats/thread
    const int bk   = tid >> 2, bnoff = (tid & 3) * 16;   // B: 16 floats/thread

    auto stage = [&](int buf, int k0) {
        const float4* ap = (const float4*)(X + (row0 + arow) * EMBED + k0 + akoff);
        _Float16* ad = As[buf] + arow * LDSPAD + akoff;
        *(v8h*)(ad)     = cvt8(ap[0], ap[1]);
        *(v8h*)(ad + 8) = cvt8(ap[2], ap[3]);

        const float4* bp = (const float4*)(W + (k0 + bk) * EMBED + n0 + bnoff);
        float4 g0 = bp[0], g1 = bp[1], g2 = bp[2], g3 = bp[3];
        float t[16] = {g0.x,g0.y,g0.z,g0.w, g1.x,g1.y,g1.z,g1.w,
                       g2.x,g2.y,g2.z,g2.w, g3.x,g3.y,g3.z,g3.w};
#pragma unroll
        for (int e = 0; e < 16; ++e)
            Bs[buf][(bnoff + e) * LDSPAD + bk] = (_Float16)t[e];
    };

    v8f c00 = zero8(), c01 = zero8(), c10 = zero8(), c11 = zero8();

    stage(0, 0);
    const int NSTEP = EMBED / 32;
    for (int i = 0; i < NSTEP; ++i) {
        __syncthreads();
        if (i + 1 < NSTEP) stage((i + 1) & 1, (i + 1) * 32);

        const _Float16* Ab = As[i & 1];
        const _Float16* Bb = Bs[i & 1];
        const v16h a0 = lds_read_a(Ab + (wm * 32 +  0 + nl) * LDSPAD + half * 8);
        const v16h a1 = lds_read_a(Ab + (wm * 32 + 16 + nl) * LDSPAD + half * 8);
        const v16h b0 = lds_read_b(Bb + (wn * 32 +  0 + nl) * LDSPAD + half * 8);
        const v16h b1 = lds_read_b(Bb + (wn * 32 + 16 + nl) * LDSPAD + half * 8);

        c00 = __builtin_amdgcn_wmma_f32_16x16x32_f16(false, a0, false, b0, (short)0, c00, false, false);
        c01 = __builtin_amdgcn_wmma_f32_16x16x32_f16(false, a0, false, b1, (short)0, c01, false, false);
        c10 = __builtin_amdgcn_wmma_f32_16x16x32_f16(false, a1, false, b0, (short)0, c10, false, false);
        c11 = __builtin_amdgcn_wmma_f32_16x16x32_f16(false, a1, false, b1, (short)0, c11, false, false);
    }

    // epilogue: bias, f16, head-interleaved [B,H,S,Dh]  (32-bit index math)
#pragma unroll
    for (int j = 0; j < 2; ++j) {
        const int gn = n0 + wn * 32 + j * 16 + nl;
        const int h = gn >> 6, dh = gn & 63;
        const float bb = bias[gn];
        const v8f* cj0 = (j == 0) ? &c00 : &c01;
        const v8f* cj1 = (j == 0) ? &c10 : &c11;
#pragma unroll
        for (int r = 0; r < 8; ++r) {
            int gm0 = row0 + wm * 32 + half * 8 + r;
            int gm1 = gm0 + 16;
            int i0 = (((gm0 >> 11) * NHEAD + h) * SEQ + (gm0 & (SEQ - 1))) * HDIM + dh;
            int i1 = (((gm1 >> 11) * NHEAD + h) * SEQ + (gm1 & (SEQ - 1))) * HDIM + dh;
            Out[i0] = (_Float16)((*cj0)[r] + bb);
            Out[i1] = (_Float16)((*cj1)[r] + bb);
        }
    }
}

// ---------------------------------------------------------------------------
// Stage 2: streaming (flash) attention per (b,h).  64 q rows/block, 16/wave.
// 32-key tiles, double-buffered: K tile via async global->LDS DMA, V tile
// transposed to n-major, P round-trips LDS for the C->A lane transpose.
// ---------------------------------------------------------------------------
__global__ __launch_bounds__(128) void attention(
    const _Float16* __restrict__ Qh, const _Float16* __restrict__ Kh,
    const _Float16* __restrict__ Vh, _Float16* __restrict__ Oh) {
    __shared__ __align__(16) _Float16 Kt[2][32 * KPAD];        // [key][dh]
    __shared__ __align__(16) _Float16 Vt[2][64 * LDSPAD];      // [dh][key]
    __shared__ __align__(16) _Float16 Pt[4][16 * LDSPAD];      // [row][key], per-wave

    const int lane = threadIdx.x, wy = threadIdx.y;
    const int half = lane >> 4, nl = lane & 15;
    const int bh = blockIdx.y;                 // b*16 + h
    const int b  = bh >> 4, h = bh & 15;
    const int qrow0 = blockIdx.x * 64 + wy * 16;
    const int tid = wy * 32 + lane;

    const _Float16* Qb = Qh + (size_t)bh * SEQ * HDIM;
    const _Float16* Kb = Kh + (size_t)bh * SEQ * HDIM;
    const _Float16* Vb = Vh + (size_t)bh * SEQ * HDIM;

    v16h qa0, qa1;
    {
        const _Float16* qp = Qb + (qrow0 + nl) * HDIM;
        qa0 = cat8(*(const v8h*)(qp + half * 8),      *(const v8h*)(qp + 16 + half * 8));
        qa1 = cat8(*(const v8h*)(qp + 32 + half * 8), *(const v8h*)(qp + 48 + half * 8));
    }

    v8f o0 = zero8(), o1 = zero8(), o2 = zero8(), o3 = zero8();
    float mI[8], lI[8];
#pragma unroll
    for (int r = 0; r < 8; ++r) { mI[r] = -1e30f; lI[r] = 0.0f; }
    const float scale = 0.125f;                // 1/sqrt(64)

    const int crow = tid >> 2;                 // 0..31 (key row)
    const int ccol = (tid & 3) * 16;           // 0/16/32/48 (dh)

    auto stage = [&](int buf, int kt) {
        const _Float16* ksrc = Kb + (kt + crow) * HDIM + ccol;
        _Float16* kdst = Kt[buf] + crow * KPAD + ccol;
#if USE_ASYNC
        ASYNC_B128(ksrc, kdst, 0);
        ASYNC_B128(ksrc, kdst, 16);
#else
        ((uint4*)kdst)[0] = ((const uint4*)ksrc)[0];
        ((uint4*)kdst)[1] = ((const uint4*)ksrc)[1];
#endif
        const _Float16* vsrc = Vb + (kt + crow) * HDIM + ccol;
#pragma unroll
        for (int e = 0; e < 16; ++e)
            Vt[buf][(ccol + e) * LDSPAD + crow] = vsrc[e];
    };

    stage(0, 0);
    const int NSTEP = SEQ / 32;
    for (int it = 0; it < NSTEP; ++it) {
        ASYNC_WAIT();
        __syncthreads();
        if (it + 1 < NSTEP) stage((it + 1) & 1, (it + 1) * 32);

        const _Float16* Ktb = Kt[it & 1];
        const _Float16* Vtb = Vt[it & 1];

        // S = Q K^T : two 16-key groups, Dh = 64 in two windows
        v8f s0 = zero8(), s1 = zero8();
#pragma unroll
        for (int g = 0; g < 2; ++g) {
            const _Float16* kp = Ktb + (g * 16 + nl) * KPAD;
            v16h kb0 = lds_read_b(kp + half * 8);
            v16h kb1 = lds_read_b(kp + 32 + half * 8);
            if (g == 0) {
                s0 = __builtin_amdgcn_wmma_f32_16x16x32_f16(false, qa0, false, kb0, (short)0, s0, false, false);
                s0 = __builtin_amdgcn_wmma_f32_16x16x32_f16(false, qa1, false, kb1, (short)0, s0, false, false);
            } else {
                s1 = __builtin_amdgcn_wmma_f32_16x16x32_f16(false, qa0, false, kb0, (short)0, s1, false, false);
                s1 = __builtin_amdgcn_wmma_f32_16x16x32_f16(false, qa1, false, kb1, (short)0, s1, false, false);
            }
        }

        // online softmax: rows 8*half+r live in lanes [half*16 .. +15]
#pragma unroll
        for (int r = 0; r < 8; ++r) {
            float x0 = s0[r] * scale, x1 = s1[r] * scale;
            float t = fmaxf(x0, x1);
#pragma unroll
            for (int mk = 1; mk <= 8; mk <<= 1) t = fmaxf(t, __shfl_xor(t, mk, 32));
            float mn = fmaxf(mI[r], t);
            float al = __expf(mI[r] - mn);
            mI[r] = mn;
            float p0 = __expf(x0 - mn), p1 = __expf(x1 - mn);
            float rs = p0 + p1;
#pragma unroll
            for (int mk = 1; mk <= 8; mk <<= 1) rs += __shfl_xor(rs, mk, 32);
            lI[r] = lI[r] * al + rs;
            o0[r] *= al; o1[r] *= al; o2[r] *= al; o3[r] *= al;
            Pt[wy][(half * 8 + r) * LDSPAD + nl]      = (_Float16)p0;
            Pt[wy][(half * 8 + r) * LDSPAD + 16 + nl] = (_Float16)p1;
        }

        // same-wave LDS RAW: DS ops retire in order per wave; force the wait
        asm volatile("s_wait_dscnt 0" ::: "memory");

        const v16h pa = lds_read_a(Pt[wy] + nl * LDSPAD + half * 8);

#pragma unroll
        for (int t = 0; t < 4; ++t) {
            v16h vb = lds_read_b(Vtb + (t * 16 + nl) * LDSPAD + half * 8);
            v8f& od = (t == 0) ? o0 : (t == 1) ? o1 : (t == 2) ? o2 : o3;
            od = __builtin_amdgcn_wmma_f32_16x16x32_f16(false, pa, false, vb, (short)0, od, false, false);
        }
    }

    // epilogue: O /= l, f16 to [B*S, EMBED] (heads concatenated)
#pragma unroll
    for (int r = 0; r < 8; ++r) {
        int s = qrow0 + half * 8 + r;
        float inv = 1.0f / lI[r];
        int base = (b * SEQ + s) * EMBED + h * HDIM + nl;
        Oh[base +  0] = (_Float16)(o0[r] * inv);
        Oh[base + 16] = (_Float16)(o1[r] * inv);
        Oh[base + 32] = (_Float16)(o2[r] * inv);
        Oh[base + 48] = (_Float16)(o3[r] * inv);
    }
}

// ---------------------------------------------------------------------------
// Stage 3: output projection.  d_out = O(f16) @ Wo(f32) + bo  (fp32 out).
// A-tile copy goes through async global->LDS DMA (pure f16 copy).
// ---------------------------------------------------------------------------
__global__ __launch_bounds__(128) void out_gemm(
    const _Float16* __restrict__ A, const float* __restrict__ W,
    const float* __restrict__ bias, float* __restrict__ Out) {
    __shared__ __align__(16) _Float16 As[2][64 * LDSPAD];
    __shared__ __align__(16) _Float16 Bs[2][64 * LDSPAD];

    const int lane = threadIdx.x, wy = threadIdx.y;
    const int half = lane >> 4, nl = lane & 15;
    const int wm = wy >> 1, wn = wy & 1;
    const int row0 = blockIdx.y * 64, n0 = blockIdx.x * 64;
    const int tid = wy * 32 + lane;

    const int arow = tid >> 1, akoff = (tid & 1) * 16;
    const int bk   = tid >> 2, bnoff = (tid & 3) * 16;

    auto stage = [&](int buf, int k0) {
        const _Float16* asrc = A + (row0 + arow) * EMBED + k0 + akoff;
        _Float16* adst = As[buf] + arow * LDSPAD + akoff;
#if USE_ASYNC
        ASYNC_B128(asrc, adst, 0);
        ASYNC_B128(asrc, adst, 16);
#else
        ((uint4*)adst)[0] = ((const uint4*)asrc)[0];
        ((uint4*)adst)[1] = ((const uint4*)asrc)[1];
#endif
        const float4* bp = (const float4*)(W + (k0 + bk) * EMBED + n0 + bnoff);
        float4 g0 = bp[0], g1 = bp[1], g2 = bp[2], g3 = bp[3];
        float t[16] = {g0.x,g0.y,g0.z,g0.w, g1.x,g1.y,g1.z,g1.w,
                       g2.x,g2.y,g2.z,g2.w, g3.x,g3.y,g3.z,g3.w};
#pragma unroll
        for (int e = 0; e < 16; ++e)
            Bs[buf][(bnoff + e) * LDSPAD + bk] = (_Float16)t[e];
    };

    v8f c00 = zero8(), c01 = zero8(), c10 = zero8(), c11 = zero8();

    stage(0, 0);
    const int NSTEP = EMBED / 32;
    for (int i = 0; i < NSTEP; ++i) {
        ASYNC_WAIT();
        __syncthreads();
        if (i + 1 < NSTEP) stage((i + 1) & 1, (i + 1) * 32);

        const _Float16* Ab = As[i & 1];
        const _Float16* Bb = Bs[i & 1];
        const v16h a0 = lds_read_a(Ab + (wm * 32 +  0 + nl) * LDSPAD + half * 8);
        const v16h a1 = lds_read_a(Ab + (wm * 32 + 16 + nl) * LDSPAD + half * 8);
        const v16h b0 = lds_read_b(Bb + (wn * 32 +  0 + nl) * LDSPAD + half * 8);
        const v16h b1 = lds_read_b(Bb + (wn * 32 + 16 + nl) * LDSPAD + half * 8);

        c00 = __builtin_amdgcn_wmma_f32_16x16x32_f16(false, a0, false, b0, (short)0, c00, false, false);
        c01 = __builtin_amdgcn_wmma_f32_16x16x32_f16(false, a0, false, b1, (short)0, c01, false, false);
        c10 = __builtin_amdgcn_wmma_f32_16x16x32_f16(false, a1, false, b0, (short)0, c10, false, false);
        c11 = __builtin_amdgcn_wmma_f32_16x16x32_f16(false, a1, false, b1, (short)0, c11, false, false);
    }

#pragma unroll
    for (int j = 0; j < 2; ++j) {
        const int gn = n0 + wn * 32 + j * 16 + nl;
        const float bb = bias[gn];
        const v8f* cj0 = (j == 0) ? &c00 : &c01;
        const v8f* cj1 = (j == 0) ? &c10 : &c11;
#pragma unroll
        for (int r = 0; r < 8; ++r) {
            int gm0 = row0 + wm * 32 + half * 8 + r;
            int gm1 = gm0 + 16;
            Out[gm0 * EMBED + gn] = (*cj0)[r] + bb;
            Out[gm1 * EMBED + gn] = (*cj1)[r] + bb;
        }
    }
}

// ---------------------------------------------------------------------------
extern "C" void kernel_launch(void* const* d_in, const int* in_sizes, int n_in,
                              void* d_out, int out_size, void* d_ws, size_t ws_size,
                              hipStream_t stream) {
    const float* X  = (const float*)d_in[0];
    const float* Wq = (const float*)d_in[1];
    const float* bq = (const float*)d_in[2];
    const float* Wk = (const float*)d_in[3];
    const float* bk = (const float*)d_in[4];
    const float* Wv = (const float*)d_in[5];
    const float* bv = (const float*)d_in[6];
    const float* Wo = (const float*)d_in[7];
    const float* bo = (const float*)d_in[8];

    const size_t elems = (size_t)ROWS * EMBED;       // 4M f16 = 8 MB each
    _Float16* Qh = (_Float16*)d_ws;
    _Float16* Kh = Qh + elems;
    _Float16* Vh = Kh + elems;
    _Float16* Oh = Vh + elems;

    dim3 blk(32, 4);
    dim3 gGemm(EMBED / 64, ROWS / 64);               // (16, 64)
    dim3 gAttn(SEQ / 64, BATCH * NHEAD);             // (32, 32)

    qkv_gemm<<<gGemm, blk, 0, stream>>>(X, Wq, bq, Qh);
    qkv_gemm<<<gGemm, blk, 0, stream>>>(X, Wk, bk, Kh);
    qkv_gemm<<<gGemm, blk, 0, stream>>>(X, Wv, bv, Vh);
    attention<<<gAttn, blk, 0, stream>>>(Qh, Kh, Vh, Oh);
    out_gemm<<<gGemm, blk, 0, stream>>>(Oh, Wo, bo, (float*)d_out);
}
